// LRU_82858509074594
// MI455X (gfx1250) — matrix operand order repaired and checked by reading, not compiled, not run on hardware
//
#include <hip/hip_runtime.h>
#include <hip/hip_bf16.h>

// LRU forward for MI455X (gfx1250): bf16 WMMA GEMMs with async-to-LDS
// double-buffered staging + fused Lambda/cumsum.  B=8, L=2048, H=1024, N=512.

typedef __bf16 bf16;
typedef __attribute__((ext_vector_type(16))) __bf16 v16bf;
typedef __attribute__((ext_vector_type(8)))  float  v8f;

#define B_SZ  8
#define LSEQ  2048
#define HDIM  1024
#define NDIM  512
#define M_TOT (B_SZ * LSEQ)   // 16384
#define K_TOT 1024            // both GEMMs: K = 1024
#define N_TOT 1024            // both GEMMs: N = 1024 (fused real|imag)

// ---------------------------------------------------------------------------
// GEMM: C(M_TOT x 1024) = A(M_TOT x 1024, bf16 row-major)
//                       x BT(1024 x 1024, bf16, stored as BT[n][k])
// Block tile 128x128, K-tile 64, 8 waves (wave32), each wave 64x32 via
// 4x2 v_wmma_f32_16x16x32_bf16 tiles. Tiles staged with
// global_load_async_to_lds_b128 (ASYNCcnt) into a double-buffered LDS.
// ---------------------------------------------------------------------------
__global__ __launch_bounds__(256)
void lru_gemm_bf16(const bf16* __restrict__ A, const bf16* __restrict__ BT,
                   float* __restrict__ C) {
    // +8 bf16 pad (16B) per row: rotates bank start by 4 banks per row so
    // 16 lanes' ds_load_b128 of 16 different rows hit disjoint banks.
    __shared__ __align__(16) bf16 lA[2][128][72];
    __shared__ __align__(16) bf16 lB[2][128][72];

    const int t    = threadIdx.x;
    const int lane = t & 31;
    const int lr   = lane & 15;   // row (A) / col (B) within 16x16 tile
    const int lh   = lane >> 4;   // K-half selector per ISA VGPR layout
    const int wave = t >> 5;
    const int wm   = (wave >> 2) * 64;  // wave M offset in block: 0 / 64
    const int wn   = (wave & 3)  * 32;  // wave N offset in block: 0/32/64/96
    const int bm   = blockIdx.y * 128;
    const int bn   = blockIdx.x * 128;

    const int rr = t >> 3;        // 0..31: staging row group
    const int cc = (t & 7) * 8;   // staging col (bf16 elems), 16B granules

    v8f acc[4][2];
#pragma unroll
    for (int m = 0; m < 4; ++m)
#pragma unroll
        for (int n = 0; n < 2; ++n)
            acc[m][n] = v8f{};

    // Stage one 128x64 bf16 tile pair into LDS buffer `buf` via async DMA.
    // GVS mode: s[base] + vgpr_byte_offset (< 2^31 here). LDS byte address is
    // the low 32 bits of the generic shared-memory pointer.
    auto stage = [&](int buf, int k0) {
#pragma unroll
        for (int p = 0; p < 4; ++p) {
            const int r = p * 32 + rr;
            const unsigned la = (unsigned)(uintptr_t)&lA[buf][r][cc];
            const unsigned lb = (unsigned)(uintptr_t)&lB[buf][r][cc];
            const unsigned ga = (unsigned)(((bm + r) * K_TOT + k0 + cc) * 2);
            const unsigned gb = (unsigned)(((bn + r) * K_TOT + k0 + cc) * 2);
            asm volatile("global_load_async_to_lds_b128 %0, %1, %2"
                         :: "v"(la), "v"(ga), "s"(A) : "memory");
            asm volatile("global_load_async_to_lds_b128 %0, %1, %2"
                         :: "v"(lb), "v"(gb), "s"(BT) : "memory");
        }
    };

    stage(0, 0);
    asm volatile("s_wait_asynccnt 0x0" ::: "memory");
    __syncthreads();

    for (int k0 = 0; k0 < K_TOT; k0 += 64) {
        const int buf = (k0 >> 6) & 1;
        if (k0 + 64 < K_TOT)            // overlap next-tile DMA with compute
            stage(buf ^ 1, k0 + 64);

#pragma unroll
        for (int kk = 0; kk < 64; kk += 32) {
            // A fragment (16x32, MxK): lane lr = row M, lh picks K-half.
            //   elems 0..7  = K[kk + lh*8 .. +7], elems 8..15 = K[kk+16+lh*8 ..]
            union { uint4 q[2]; v16bf v; } a[4], b[2];
#pragma unroll
            for (int m = 0; m < 4; ++m) {
                const bf16* p = &lA[buf][wm + m * 16 + lr][kk + lh * 8];
                a[m].q[0] = *(const uint4*)(p);
                a[m].q[1] = *(const uint4*)(p + 16);
            }
            // B fragment (32x16, KxN): lane lr = col N, same K order as A:
            //   elems 0..7 = K[kk + lh*8 ..], elems 8..15 = K[kk+16+lh*8 ..].
#pragma unroll
            for (int n = 0; n < 2; ++n) {
                const bf16* p = &lB[buf][wn + n * 16 + lr][kk + lh * 8];
                b[n].q[0] = *(const uint4*)(p);
                b[n].q[1] = *(const uint4*)(p + 16);
            }
#pragma unroll
            for (int m = 0; m < 4; ++m)
#pragma unroll
                for (int n = 0; n < 2; ++n)
                    acc[m][n] = __builtin_amdgcn_wmma_f32_16x16x32_bf16(
                        false, a[m].v, false, b[n].v, (short)0, acc[m][n],
                        false, false);
        }

        // My async DMAs into buf^1 must land, and everyone must be done
        // reading buf, before the next iteration touches either.
        asm volatile("s_wait_asynccnt 0x0" ::: "memory");
        __syncthreads();
    }

    // C/D layout: lane 0-15 -> N=lr, rows M = vgpr r (+8 for lanes 16-31).
#pragma unroll
    for (int m = 0; m < 4; ++m)
#pragma unroll
        for (int n = 0; n < 2; ++n) {
            const int col = bn + wn + n * 16 + lr;
#pragma unroll
            for (int r = 0; r < 8; ++r) {
                const int row = bm + wm + m * 16 + lh * 8 + r;
                C[(size_t)row * N_TOT + col] = acc[m][n][r];
            }
        }
}

// ---------------------------------------------------------------------------
// Pack kernels: build bf16 operands.
// ---------------------------------------------------------------------------
__global__ __launch_bounds__(256)
void lru_cvt_u(const float* __restrict__ u, bf16* __restrict__ ub, int n4) {
    const int i = blockIdx.x * 256 + threadIdx.x;
    if (i < n4) {
        const float4 v = ((const float4*)u)[i];
        union { bf16 h[4]; uint2 u2; } o;
        o.h[0] = (bf16)v.x; o.h[1] = (bf16)v.y;
        o.h[2] = (bf16)v.z; o.h[3] = (bf16)v.w;
        ((uint2*)ub)[i] = o.u2;
    }
}

// BT[n][h]: n<512 -> Br[h][n], else Bi[h][n-512].   (1024x1024)
__global__ __launch_bounds__(256)
void lru_pack_B(const float* __restrict__ Br, const float* __restrict__ Bi,
                bf16* __restrict__ BT) {
    const int i = blockIdx.x * 256 + threadIdx.x;  // 1M
    const int n = i >> 10;
    const int h = i & 1023;
    const float v = (n < NDIM) ? Br[h * NDIM + n] : Bi[h * NDIM + (n - NDIM)];
    BT[i] = (bf16)v;
}

// CT[h][k]: k<512 -> Cr[k][h], else -Ci[k-512][h]  (minus sign folded in).
__global__ __launch_bounds__(256)
void lru_pack_C(const float* __restrict__ Cr, const float* __restrict__ Ci,
                bf16* __restrict__ CT) {
    const int i = blockIdx.x * 256 + threadIdx.x;  // 1M
    const int h = i >> 10;
    const int k = i & 1023;
    const float v = (k < NDIM) ? Cr[k * HDIM + h] : -Ci[(k - NDIM) * HDIM + h];
    CT[i] = (bf16)v;
}

// ---------------------------------------------------------------------------
// Fused Lambda scaling + cumsum over batch (axis 0, only 8 terms).
// One thread per (l, j); emits bf16 x in GEMM-2 layout [m][j | 512+j].
// ---------------------------------------------------------------------------
__global__ __launch_bounds__(256)
void lru_lambda_cumsum(const float* __restrict__ Bu, const int* __restrict__ len,
                       const float* __restrict__ nu_log,
                       const float* __restrict__ th_log,
                       bf16* __restrict__ X) {
    const int idx = blockIdx.x * 256 + threadIdx.x;  // L*N = 1M threads
    const int j = idx & (NDIM - 1);
    const int l = idx >> 9;
    const float nu = __expf(nu_log[j]);   // -ln(lam_mod)
    const float th = __expf(th_log[j]);   // lam_arg
    float sr = 0.f, si = 0.f;
#pragma unroll
    for (int b = 0; b < B_SZ; ++b) {
        const float e   = fmaxf((float)(len[b] - (l + 1)), 0.f);
        const float mag = __expf(-e * nu);           // lam_mod^e
        float s, c;
        __sincosf(e * th, &s, &c);
        const float Lr = mag * c, Li = mag * s;
        const size_t base = (size_t)(b * LSEQ + l) * N_TOT;
        const float br = Bu[base + j];
        const float bi = Bu[base + NDIM + j];
        sr += Lr * br - Li * bi;   // running cumsum over batch
        si += Lr * bi + Li * br;
        X[base + j]        = (bf16)sr;
        X[base + NDIM + j] = (bf16)si;
    }
}

// ---------------------------------------------------------------------------
// Workspace layout (bytes):            total = 100 MB
//   [0,   32M)  u_bf16 (stage 1 A) -> reused as X (stage 2 A) after GEMM 1
//   [32M, 34M)  BT bf16
//   [34M, 36M)  CT bf16
//   [36M, 100M) Bu f32 (16384 x 1024)
// ---------------------------------------------------------------------------
extern "C" void kernel_launch(void* const* d_in, const int* in_sizes, int n_in,
                              void* d_out, int out_size, void* d_ws, size_t ws_size,
                              hipStream_t stream) {
    const float* u      = (const float*)d_in[0];
    const int*   len    = (const int*)  d_in[1];
    const float* nu_log = (const float*)d_in[2];
    const float* th_log = (const float*)d_in[3];
    const float* Br     = (const float*)d_in[4];
    const float* Bi     = (const float*)d_in[5];
    const float* Cr     = (const float*)d_in[6];
    const float* Ci     = (const float*)d_in[7];
    float* y = (float*)d_out;

    char* ws = (char*)d_ws;
    bf16*  uX = (bf16*)(ws);                                     // 32 MB
    bf16*  BT = (bf16*)(ws + (size_t)M_TOT * K_TOT * 2);         //  2 MB
    bf16*  CT = (bf16*)(ws + (size_t)M_TOT * K_TOT * 2 + 2097152);
    float* Bu = (float*)(ws + (size_t)M_TOT * K_TOT * 2 + 2 * 2097152);

    const int n4 = (M_TOT * HDIM) / 4;  // 4,194,304
    lru_cvt_u <<<(n4 + 255) / 256, 256, 0, stream>>>(u, uX, n4);
    lru_pack_B<<<(N_TOT * HDIM) / 256, 256, 0, stream>>>(Br, Bi, BT);
    lru_pack_C<<<(N_TOT * HDIM) / 256, 256, 0, stream>>>(Cr, Ci, CT);

    dim3 gemm_grid(N_TOT / 128, M_TOT / 128);  // (8, 128)
    lru_gemm_bf16<<<gemm_grid, 256, 0, stream>>>(uX, BT, Bu);

    lru_lambda_cumsum<<<(LSEQ * NDIM) / 256, 256, 0, stream>>>(
        Bu, len, nu_log, th_log, uX /* X reuses u_bf16 slot */);

    lru_gemm_bf16<<<gemm_grid, 256, 0, stream>>>(uX, CT, y);
}
